// VanillaRNN_30940944400985
// MI455X (gfx1250) — matrix operand order repaired and checked by reading, not compiled, not run on hardware
//
#include <hip/hip_runtime.h>
#include <math.h>

// ---------------------------------------------------------------------------
// Vanilla RNN on MI455X (gfx1250), wave32 + v_wmma_f32_16x16x32_f16.
//
//   h_{t+1} = tanh(Whh @ h_t + Whx * x[:,t] + bh)   (511 steps, H=1024, B=256)
//   out     = (Wph @ h_T + bp)^T                    ([B, C] f32)
//
// v2: the run is L2-bandwidth bound (Whh = 2MB f16 lives in the 192MB L2
// across all 511 steps; HBM is touched once). To cut L2->WGP traffic, the
// step kernel now uses a 128x64 workgroup tile: each 32-wide K-chunk of A
// (128x32) and B (32x64) is staged into LDS once (double-buffered, 24KB of
// the 320KB WGP LDS) and re-read by all 8 waves via ds_load_b128. Per-step
// cache traffic drops from ~32MB to ~12MB. Each wave computes a 32x32 block
// (2x2 WMMA tiles) with f32 accumulation.
// ---------------------------------------------------------------------------

typedef __attribute__((ext_vector_type(16))) _Float16 v16h;
typedef __attribute__((ext_vector_type(8)))  _Float16 v8h;
typedef __attribute__((ext_vector_type(4)))  _Float16 v4h;
typedef __attribute__((ext_vector_type(8)))  float    v8f;
typedef __attribute__((ext_vector_type(4)))  float    v4f;

constexpr int Bn = 256;   // batch
constexpr int Tn = 512;   // sequence length
constexpr int Hn = 1024;  // hidden size
constexpr int Cn = 1024;  // output classes

union HFrag { v16h v; v8h h[2]; };

#define WMMA_F16(a, b, c) \
  __builtin_amdgcn_wmma_f32_16x16x32_f16(false, (a), false, (b), (short)0, (c), false, false)

// A fragment from a row-major f16 source with leading dim `ld`.
// ISA 16-bit A 16x32 layout: lane L holds row M = L%16; lo half-wave holds
// K = {k..k+7, k+16..k+23}, hi half-wave K = {k+8..k+15, k+24..k+31}.
__device__ __forceinline__ v16h load_a_frag(const _Float16* __restrict__ A,
                                            int m, int k, int ld, int hi) {
  HFrag f;
  const _Float16* p = A + (size_t)m * ld + k + hi * 8;
  f.h[0] = *(const v8h*)p;
  f.h[1] = *(const v8h*)(p + 16);
  return f.v;
}

// B fragment from a K-contiguous (per column) f16 source, leading dim `ld`.
// Lanes 0-15 hold K = {k..k+7, k+16..k+23} of column n = L%16, lanes 16-31 K = {k+8..k+15, k+24..k+31}.
__device__ __forceinline__ v16h load_b_frag(const _Float16* __restrict__ Bm,
                                            int k, int n, int ld, int hi) {
  HFrag f;
  const _Float16* p = Bm + (size_t)n * ld + k + hi * 8;
  f.h[0] = *(const v8h*)p;
  f.h[1] = *(const v8h*)(p + 16);
  return f.v;
}

// ---------------------------------------------------------------------------
// One recurrence step: hnext = tanh(Whh16 @ hcur + Whx * x[:,t] + bh)
// Grid: 32 blocks x 256 threads (8 waves). Block tile 128M x 64N; wave tile
// 32x32 (2x2 WMMA). A/B K-chunks double-buffered through LDS.
// ---------------------------------------------------------------------------
__global__ __launch_bounds__(256)
void rnn_step_wmma(const _Float16* __restrict__ Whh16,  // [H][H] row-major
                   const _Float16* __restrict__ hcur,   // [B] cols of H f16
                   const float*    __restrict__ x,      // [B][T] row-major
                   const float*    __restrict__ Whx,    // [H]
                   const float*    __restrict__ bh,     // [H]
                   _Float16*       __restrict__ hnext,  // [B] cols of H f16
                   int t) {
  __shared__ alignas(16) _Float16 sA[2][128][32];  // 2 x 8KB
  __shared__ alignas(16) _Float16 sB[2][64][32];   // 2 x 4KB

  const int tid  = threadIdx.x;
  const int lane = tid & 31;
  const int lsub = lane & 15;
  const int hi   = lane >> 4;
  const int w    = tid >> 5;          // wave 0..7
  const int wm   = (w & 3) * 32;      // wave M offset inside block tile
  const int wn   = (w >> 2) * 32;     // wave N offset inside block tile

  const int blockM = (blockIdx.x & 7) * 128;   // 8 M blocks over H
  const int blockN = (blockIdx.x >> 3) * 64;   // 4 N blocks over B

  // Cooperative staging map: A = 128 rows x 32 halfs (8KB -> 32B/thread),
  // B = 64 cols x 32 halfs (4KB -> 16B/thread).
  const int rA = tid >> 1;
  const int qA = (tid & 1) * 16;
  const _Float16* gA = Whh16 + (size_t)(blockM + rA) * Hn + qA;
  const int cB = tid >> 2;
  const int qB = (tid & 3) * 8;
  const _Float16* gB = hcur + (size_t)(blockN + cB) * Hn + qB;

  constexpr int NC = Hn / 32;  // 32 K-chunks

  auto stage = [&](int kc, int buf) {
    v8h a0 = *(const v8h*)(gA + kc * 32);
    v8h a1 = *(const v8h*)(gA + kc * 32 + 8);
    v8h b0 = *(const v8h*)(gB + kc * 32);
    *(v8h*)(&sA[buf][rA][qA])     = a0;
    *(v8h*)(&sA[buf][rA][qA + 8]) = a1;
    *(v8h*)(&sB[buf][cB][qB])     = b0;
    if (kc + 2 < NC) {  // keep the chunk after next in flight toward WGP$
      __builtin_prefetch((const void*)(gA + (kc + 2) * 32), 0, 3);
      __builtin_prefetch((const void*)(gB + (kc + 2) * 32), 0, 3);
    }
  };

  v8f acc00 = {}, acc01 = {}, acc10 = {}, acc11 = {};

  stage(0, 0);
  for (int kc = 0; kc < NC; ++kc) {
    __syncthreads();                  // buf[kc&1] ready; prior reads of the
    const int buf = kc & 1;           // other buffer finished last iteration
    if (kc + 1 < NC) stage(kc + 1, buf ^ 1);

    HFrag a0, a1, b0, b1;
    {
      const _Float16* p = &sA[buf][wm + lsub][hi * 8];
      a0.h[0] = *(const v8h*)p;  a0.h[1] = *(const v8h*)(p + 16);
      p = &sA[buf][wm + 16 + lsub][hi * 8];
      a1.h[0] = *(const v8h*)p;  a1.h[1] = *(const v8h*)(p + 16);
      p = &sB[buf][wn + lsub][hi * 8];
      b0.h[0] = *(const v8h*)p;  b0.h[1] = *(const v8h*)(p + 16);
      p = &sB[buf][wn + 16 + lsub][hi * 8];
      b1.h[0] = *(const v8h*)p;  b1.h[1] = *(const v8h*)(p + 16);
    }
    acc00 = WMMA_F16(a0.v, b0.v, acc00);
    acc01 = WMMA_F16(a0.v, b1.v, acc01);
    acc10 = WMMA_F16(a1.v, b0.v, acc10);
    acc11 = WMMA_F16(a1.v, b1.v, acc11);
  }

  // Epilogue: D layout -> lane has column n, rows m0+hi*8..+7 in vgprs.
  // Bias + input term + tanh, f16 convert, one aligned b128 store per tile;
  // layout matches the next step's B-fragment loads exactly.
  auto emit = [&](v8f acc, int mTile, int nTile) {
    const int n  = nTile + lsub;
    const int m0 = mTile + hi * 8;
    const float xt = x[(size_t)n * Tn + t];
    union { v8h v; _Float16 e[8]; } o;
#pragma unroll
    for (int v = 0; v < 8; ++v) {
      float z = acc[v] + Whx[m0 + v] * xt + bh[m0 + v];
      o.e[v] = (_Float16)tanhf(z);
    }
    *(v8h*)(hnext + (size_t)n * Hn + m0) = o.v;
  };
  emit(acc00, blockM + wm,      blockN + wn);
  emit(acc01, blockM + wm,      blockN + wn + 16);
  emit(acc10, blockM + wm + 16, blockN + wn);
  emit(acc11, blockM + wm + 16, blockN + wn + 16);
}

// ---------------------------------------------------------------------------
// Projection (runs once): out[b][c] = (Wph @ h)[c][b] + bp[c], f32 [B][C].
// Register-only 2x2 WMMA tiles per wave; 64 blocks x 128 threads.
// ---------------------------------------------------------------------------
__global__ __launch_bounds__(128)
void rnn_proj_wmma(const _Float16* __restrict__ Wph16,  // [C][H] row-major
                   const _Float16* __restrict__ h,      // [B] cols of H f16
                   const float*    __restrict__ bp,     // [C]
                   float*          __restrict__ out) {  // [B][C]
  const int lane = threadIdx.x & 31;
  const int lsub = lane & 15;
  const int hi   = lane >> 4;
  const int gw   = blockIdx.x * 4 + (threadIdx.x >> 5);
  const int mBase = (gw & 31) * 32;   // 32 row blocks over C
  const int nBase = (gw >> 5) * 32;   // 8 col blocks over B

  v8f acc00 = {}, acc01 = {}, acc10 = {}, acc11 = {};

#pragma unroll 4
  for (int k = 0; k < Hn; k += 32) {
    v16h a0 = load_a_frag(Wph16, mBase +      lsub, k, Hn, hi);
    v16h a1 = load_a_frag(Wph16, mBase + 16 + lsub, k, Hn, hi);
    v16h b0 = load_b_frag(h, k, nBase +      lsub, Hn, hi);
    v16h b1 = load_b_frag(h, k, nBase + 16 + lsub, Hn, hi);
    acc00 = WMMA_F16(a0, b0, acc00);
    acc01 = WMMA_F16(a0, b1, acc01);
    acc10 = WMMA_F16(a1, b0, acc10);
    acc11 = WMMA_F16(a1, b1, acc11);
  }

  auto emit = [&](v8f acc, int mTile, int nTile) {
    const int n  = nTile + lsub;       // batch index
    const int m0 = mTile + hi * 8;     // class index base
    v4f s0, s1;
#pragma unroll
    for (int v = 0; v < 4; ++v) {
      s0[v] = acc[v]     + bp[m0 + v];
      s1[v] = acc[4 + v] + bp[m0 + 4 + v];
    }
    *(v4f*)(out + (size_t)n * Cn + m0)     = s0;
    *(v4f*)(out + (size_t)n * Cn + m0 + 4) = s1;
  };
  emit(acc00, mBase,      nBase);
  emit(acc01, mBase,      nBase + 16);
  emit(acc10, mBase + 16, nBase);
  emit(acc11, mBase + 16, nBase + 16);
}

// f32 -> f16 weight conversion, 4 elements/thread (b128 load, b64 store).
__global__ void cvt_f32_to_f16(const float* __restrict__ src,
                               _Float16* __restrict__ dst, int n) {
  int i = (blockIdx.x * blockDim.x + threadIdx.x) * 4;
  if (i + 3 < n) {
    float4 f = *(const float4*)(src + i);
    union { v4h v; _Float16 e[4]; } o;
    o.e[0] = (_Float16)f.x; o.e[1] = (_Float16)f.y;
    o.e[2] = (_Float16)f.z; o.e[3] = (_Float16)f.w;
    *(v4h*)(dst + i) = o.v;
  }
}

__global__ void zero_b32(unsigned int* __restrict__ p, int n32) {
  int i = blockIdx.x * blockDim.x + threadIdx.x;
  if (i < n32) p[i] = 0u;
}

extern "C" void kernel_launch(void* const* d_in, const int* in_sizes, int n_in,
                              void* d_out, int out_size, void* d_ws, size_t ws_size,
                              hipStream_t stream) {
  const float* x   = (const float*)d_in[0];  // [B, T]
  const float* Whx = (const float*)d_in[1];  // [H, 1]
  const float* Whh = (const float*)d_in[2];  // [H, H]
  const float* Wph = (const float*)d_in[3];  // [C, H]
  const float* bh  = (const float*)d_in[4];  // [H]
  const float* bp  = (const float*)d_in[5];  // [C]
  float* out = (float*)d_out;                // [B, C]

  char* ws = (char*)d_ws;
  _Float16* Whh16 = (_Float16*)(ws);                                   // 2 MB
  _Float16* Wph16 = (_Float16*)(ws + (size_t)Hn * Hn * 2);             // 2 MB
  _Float16* hA    = (_Float16*)(ws + (size_t)Hn * Hn * 4);             // 512 KB
  _Float16* hB    = (_Float16*)(ws + (size_t)Hn * Hn * 4
                                   + (size_t)Bn * Hn * 2);             // 512 KB

  // Weight conversion (re-done every call; inputs are never mutated).
  cvt_f32_to_f16<<<dim3((Hn * Hn) / (256 * 4)), 256, 0, stream>>>(Whh, Whh16, Hn * Hn);
  cvt_f32_to_f16<<<dim3((Cn * Hn) / (256 * 4)), 256, 0, stream>>>(Wph, Wph16, Cn * Hn);

  // h0 = 0 (f16, B columns of H).
  zero_b32<<<dim3((Bn * Hn / 2) / 256), 256, 0, stream>>>((unsigned int*)hA, Bn * Hn / 2);

  // 511 recurrence steps, ping-ponging h buffers.
  _Float16* cur = hA;
  _Float16* nxt = hB;
  for (int t = 0; t < Tn - 1; ++t) {
    rnn_step_wmma<<<dim3(32), 256, 0, stream>>>(Whh16, cur, x, Whx, bh, nxt, t);
    _Float16* tmp = cur; cur = nxt; nxt = tmp;
  }

  // Final projection to [B, C] f32.
  rnn_proj_wmma<<<dim3(64), 128, 0, stream>>>(Wph16, cur, bp, out);
}
